// LoRALinear_25185688224711
// MI455X (gfx1250) — matrix operand order repaired and checked by reading, not compiled, not run on hardware
//
#include <hip/hip_runtime.h>
#include <hip/hip_bf16.h>

typedef _Float16 v4h  __attribute__((ext_vector_type(4)));
typedef _Float16 v8h  __attribute__((ext_vector_type(8)));
typedef _Float16 v16h __attribute__((ext_vector_type(16)));
typedef float    v8f  __attribute__((ext_vector_type(8)));
typedef int      v4i  __attribute__((ext_vector_type(4)));

#define AS1 __attribute__((address_space(1)))
#define AS3 __attribute__((address_space(3)))

#define D_IN   4096
#define D_OUT  4096
#define M_TOK  8192          // B*S = 4*2048
#define KE     4128          // 4096 + 16 (lora) + 1 (rowsum) + 1 (bias) + 14 pad
#define NSTAGE (KE / 32)     // 129 K-stages of 32 halves
#define SROW   32            // LDS row stride in halves (64B, no pad; XOR-swizzled)
#define BM     256           // block tile M
#define BN     128           // block tile N
#define ABUFH  (BM * SROW)   // halves per A stage buffer (16 KB)
#define BBUFH  (BN * SROW)   // halves per B stage buffer (8 KB)
#define NBUF   3
#define SCALING 2.0f

#if __has_builtin(__builtin_amdgcn_global_load_async_to_lds_b128)
#define HAVE_ASYNC 1
#else
#define HAVE_ASYNC 0
#endif

#if __has_builtin(__builtin_amdgcn_s_wait_asynccnt)
#define WAIT_ASYNC(n) __builtin_amdgcn_s_wait_asynccnt(n)
#else
#define WAIT_ASYNC(n) asm volatile("s_wait_asynccnt %0" ::"i"(n))
#endif

union V16 { v16h v; struct { v8h lo, hi; } p; };

// XOR-swizzle of the four 16B segments within a 64B LDS row: makes the 16
// fragment-load lanes (rows base..base+15, fixed seg) hit disjoint banks.
__device__ __forceinline__ int lds_off(int row, int seg) {  // halves
  return row * SROW + ((seg ^ (row & 3) ^ ((row >> 2) & 3)) << 3);
}

#if HAVE_ASYNC
__device__ __forceinline__ void async128(const _Float16* g, _Float16* l) {
  // generic LDS address: low 32 bits ARE the LDS byte offset (ISA aperture rule)
  __builtin_amdgcn_global_load_async_to_lds_b128(
      (AS1 v4i*)(uintptr_t)g, (AS3 v4i*)(uint32_t)(uintptr_t)l, 0, 0);
}
#endif

// ---------------------------------------------------------------------------
// Kernel 1: extended f16 weights We[D_OUT][KE]
//   k<4096: wq (exact in f16) | 4096..4111: SCALING*lora_B/scale
//   4112: -zero | 4113: bias/scale | else 0
// ---------------------------------------------------------------------------
__global__ __launch_bounds__(256) void prep_We(
    const int* __restrict__ wq, const float* __restrict__ scale,
    const float* __restrict__ zero, const float* __restrict__ loraB,
    const float* __restrict__ bias, _Float16* __restrict__ We) {
  const int o = blockIdx.x;
  const float sc  = scale[o];
  const float inv = 1.0f / sc;
  const float zp  = zero[o];
  const float bs  = bias[o];
  const size_t wb = (size_t)o * D_IN;
  const size_t eb = (size_t)o * KE;
  for (int k = threadIdx.x; k < KE; k += 256) {
    float v;
    if (k < D_IN)            v = (float)wq[wb + k];
    else if (k < D_IN + 16)  v = SCALING * loraB[o * 16 + (k - D_IN)] * inv;
    else if (k == D_IN + 16) v = -zp;
    else if (k == D_IN + 17) v = bs * inv;
    else                     v = 0.0f;
    We[eb + k] = (_Float16)v;
  }
}

// ---------------------------------------------------------------------------
// Kernel 2: extended f16 activations Xe[M_TOK][KE]
//   k<4096: x | 4096..4111: t[m][r]=x@A^T | 4112: rowsum | 4113: 1 | else 0
// ---------------------------------------------------------------------------
__global__ __launch_bounds__(256) void prep_Xe(
    const float* __restrict__ x, const float* __restrict__ loraA,
    _Float16* __restrict__ Xe) {
  __shared__ float xrow[D_IN];
  __shared__ float red[256];
  const int tid = threadIdx.x;
  const int m = blockIdx.x;
  const size_t xb = (size_t)m * D_IN;
  const size_t eb = (size_t)m * KE;

  const float4* x4 = (const float4*)(x + xb);
  float4* xr4 = (float4*)xrow;
  float sum = 0.0f;
#pragma unroll
  for (int i = 0; i < 4; ++i) {
    int q = tid + i * 256;
    float4 v = x4[q];
    sum += v.x + v.y + v.z + v.w;
    xr4[q] = v;
    v4h h;
    h[0] = (_Float16)v.x; h[1] = (_Float16)v.y;
    h[2] = (_Float16)v.z; h[3] = (_Float16)v.w;
    *(v4h*)(Xe + eb + (size_t)q * 4) = h;
  }
  red[tid] = sum;
  __syncthreads();
  for (int st = 128; st > 0; st >>= 1) {
    if (tid < st) red[tid] += red[tid + st];
    __syncthreads();
  }
  const float rowsum = red[0];

  const int r = tid & 15, g = tid >> 4;
  const float4* A4 = (const float4*)(loraA + (size_t)r * D_IN);
  float acc = 0.0f;
#pragma unroll 4
  for (int q = g * 64; q < g * 64 + 64; ++q) {
    float4 xv = xr4[q];
    float4 av = A4[q];
    acc += xv.x * av.x + xv.y * av.y + xv.z * av.z + xv.w * av.w;
  }
  __syncthreads();
  red[r * 16 + g] = acc;
  __syncthreads();

  if (tid < 16) {
    float s = 0.0f;
#pragma unroll
    for (int j = 0; j < 16; ++j) s += red[tid * 16 + j];
    Xe[eb + D_IN + tid] = (_Float16)s;
  } else if (tid == 16) {
    Xe[eb + D_IN + 16] = (_Float16)rowsum;
  } else if (tid == 17) {
    Xe[eb + D_IN + 17] = (_Float16)1.0f;
  } else if (tid < 32) {
    Xe[eb + D_IN + tid] = (_Float16)0.0f;
  }
}

// ---------------------------------------------------------------------------
// Kernel 3: f16 WMMA GEMM  out = scale[n] * (Xe @ We^T)
// 256x128 block, 8 waves (4x2), wave tile 64x64 (4x4 accum tiles),
// 3-deep LDS ring fed by async global->LDS copies, 1 barrier/stage.
// ---------------------------------------------------------------------------
__global__ __launch_bounds__(256) void gemm_wmma(
    const _Float16* __restrict__ Xe, const _Float16* __restrict__ We,
    const float* __restrict__ scale, float* __restrict__ out) {
  __shared__ _Float16 smA[NBUF][ABUFH];   // 48 KB
  __shared__ _Float16 smB[NBUF][BBUFH];   // 24 KB

  const int tid   = threadIdx.x;
  const int lane  = tid & 31;
  const int wid   = tid >> 5;
  const int waveM = wid >> 1;          // 0..3 -> 64-row slabs
  const int waveN = wid & 1;           // 0..1 -> 64-col slabs
  const int m0 = blockIdx.y * BM;
  const int n0 = blockIdx.x * BN;

  // staging A: 1 thread per row (256 rows), all four 16B segments of the 64B row
  const _Float16* gA = Xe + (size_t)(m0 + tid) * KE;
  _Float16* sA[4];
#pragma unroll
  for (int s = 0; s < 4; ++s) sA[s] = &smA[0][lds_off(tid, s)];
  // staging B: 2 threads per row (128 rows), two 16B segments each
  const int brow = tid >> 1;
  const int bseg = (tid & 1) * 2;
  const _Float16* gB = We + (size_t)(n0 + brow) * KE + bseg * 8;
  _Float16* sB0 = &smB[0][lds_off(brow, bseg)];
  _Float16* sB1 = &smB[0][lds_off(brow, bseg + 1)];

  // fragment offsets (per-lane constants, swizzle folded in)
  const int half = lane >> 4, l15 = lane & 15;
  int aoff[4][2], boff[4][2];
#pragma unroll
  for (int tm = 0; tm < 4; ++tm) {
    int r = waveM * 64 + tm * 16 + l15;
    aoff[tm][0] = lds_off(r, half);      // K half*8 .. half*8+7
    aoff[tm][1] = lds_off(r, half + 2);  // K 16+half*8 ..
  }
#pragma unroll
  for (int tn = 0; tn < 4; ++tn) {
    int r = waveN * 64 + tn * 16 + l15;
    boff[tn][0] = lds_off(r, half);      // K half*8 .. half*8+7
    boff[tn][1] = lds_off(r, half + 2);  // K 16+half*8 ..
  }

  v8f acc[4][4] = {};

  auto compute = [&](int buf) {
    const _Float16* bA = &smA[buf][0];
    const _Float16* bB = &smB[buf][0];
    V16 af[4], bf[4];
#pragma unroll
    for (int tm = 0; tm < 4; ++tm) {
      af[tm].p.lo = *(const v8h*)(bA + aoff[tm][0]);
      af[tm].p.hi = *(const v8h*)(bA + aoff[tm][1]);
    }
#pragma unroll
    for (int tn = 0; tn < 4; ++tn) {
      bf[tn].p.lo = *(const v8h*)(bB + boff[tn][0]);
      bf[tn].p.hi = *(const v8h*)(bB + boff[tn][1]);
    }
#pragma unroll
    for (int tm = 0; tm < 4; ++tm)
#pragma unroll
      for (int tn = 0; tn < 4; ++tn)
        acc[tm][tn] = __builtin_amdgcn_wmma_f32_16x16x32_f16(
            false, af[tm].v, false, bf[tn].v, (short)0, acc[tm][tn], false, false);
  };

#if HAVE_ASYNC
  // 6 async b128 per thread per stage (A:4, B:2)
  auto issue = [&](int s, int buf) {
    const int kc = s * 32;
    const int boA = buf * ABUFH;
    const int boB = buf * BBUFH;
#pragma unroll
    for (int g = 0; g < 4; ++g) async128(gA + kc + g * 8, sA[g] + boA);
    async128(gB + kc,     sB0 + boB);
    async128(gB + kc + 8, sB1 + boB);
  };
  // pipeline: wait -> barrier -> compute(s) -> issue(s+2).
  // issue overwrites buf[(s+2)%3] == buf[(s-1)%3]; all waves finished reading
  // stage s-1 before this iteration's barrier -> single barrier per stage.
  issue(0, 0);
  issue(1, 1);
  int cb = 0;  // buffer of stage s
#pragma unroll 1
  for (int s = 0; s < NSTAGE - 2; ++s) {
    WAIT_ASYNC(6);         // stages s+1 (6) outstanding after wait -> stage s done
    __syncthreads();
    compute(cb);
    int nb = cb + 2; if (nb >= NBUF) nb -= NBUF;
    issue(s + 2, nb);
    cb = cb + 1; if (cb >= NBUF) cb = 0;
  }
  WAIT_ASYNC(6);
  __syncthreads();
  compute(cb);
  cb = cb + 1; if (cb >= NBUF) cb = 0;
  WAIT_ASYNC(0);
  __syncthreads();
  compute(cb);
#else
  // fallback: register staging, 2-barrier pipeline on the ring
  uint4 ra[4], rb[2];
  auto gload = [&](int kc) {
#pragma unroll
    for (int g = 0; g < 4; ++g) ra[g] = *(const uint4*)(gA + kc + g * 8);
    rb[0] = *(const uint4*)(gB + kc);
    rb[1] = *(const uint4*)(gB + kc + 8);
  };
  auto sstore = [&](int buf) {
#pragma unroll
    for (int g = 0; g < 4; ++g) *(uint4*)(sA[g] + buf * ABUFH) = ra[g];
    *(uint4*)(sB0 + buf * BBUFH) = rb[0];
    *(uint4*)(sB1 + buf * BBUFH) = rb[1];
  };
  gload(0);
  sstore(0);
  __syncthreads();
#pragma unroll 1
  for (int s = 1; s < NSTAGE; ++s) {
    gload(s * 32);
    compute((s - 1) % NBUF);
    __syncthreads();
    sstore(s % NBUF);
    __syncthreads();
  }
  compute((NSTAGE - 1) % NBUF);
#endif

  // epilogue: out = scale[n] * acc (bias/zero/lora folded into K)
  float scl[4];
#pragma unroll
  for (int tn = 0; tn < 4; ++tn)
    scl[tn] = scale[n0 + waveN * 64 + tn * 16 + l15];
#pragma unroll
  for (int tm = 0; tm < 4; ++tm) {
#pragma unroll
    for (int tn = 0; tn < 4; ++tn) {
      const int n = n0 + waveN * 64 + tn * 16 + l15;
#pragma unroll
      for (int r = 0; r < 8; ++r) {
        const int m = m0 + waveM * 64 + tm * 16 + r + half * 8;
        out[(size_t)m * D_OUT + n] = scl[tn] * acc[tm][tn][r];
      }
    }
  }
}

// ---------------------------------------------------------------------------
extern "C" void kernel_launch(void* const* d_in, const int* in_sizes, int n_in,
                              void* d_out, int out_size, void* d_ws, size_t ws_size,
                              hipStream_t stream) {
  const float* x     = (const float*)d_in[0];
  const int*   wq    = (const int*)d_in[1];
  const float* scale = (const float*)d_in[2];
  const float* zero  = (const float*)d_in[3];
  const float* loraA = (const float*)d_in[4];
  const float* loraB = (const float*)d_in[5];
  const float* bias  = (const float*)d_in[6];
  float* out = (float*)d_out;

  _Float16* Xe = (_Float16*)d_ws;                                   // 8192*4128*2 B
  _Float16* We = (_Float16*)((char*)d_ws + (size_t)M_TOK * KE * 2); // 4096*4128*2 B

  prep_We<<<D_OUT, 256, 0, stream>>>(wq, scale, zero, loraB, bias, We);
  prep_Xe<<<M_TOK, 256, 0, stream>>>(x, loraA, Xe);
  gemm_wmma<<<dim3(D_OUT / BN, M_TOK / BM), 256, 0, stream>>>(Xe, We, scale, out);
}
